// ClickHoldNet_89807766159582
// MI455X (gfx1250) — matrix operand order repaired and checked by reading, not compiled, not run on hardware
//
#include <hip/hip_runtime.h>
#include <hip/hip_bf16.h>

typedef __attribute__((ext_vector_type(16))) _Float16 v16h;
typedef __attribute__((ext_vector_type(8)))  _Float16 v8h;
typedef __attribute__((ext_vector_type(8)))  float    v8f;
typedef int i32x4 __attribute__((vector_size(16)));

#define HH 72
#define WW 96
#define NPIX (HH*WW)   // 6912
#define NB 256
#define CHUNK 32
#define EPS 1e-5f

#define AS1 __attribute__((address_space(1)))
#define AS3 __attribute__((address_space(3)))

__device__ __forceinline__ int iclamp(int v, int lo, int hi) {
    return v < lo ? lo : (v > hi ? hi : v);
}

// 16-byte async global->LDS copy (ASYNCcnt-tracked on CDNA5), with fallback.
__device__ __forceinline__ void async_copy16(const void* g, void* l) {
#if __has_builtin(__builtin_amdgcn_global_load_async_to_lds_b128)
    __builtin_amdgcn_global_load_async_to_lds_b128(
        (AS1 i32x4*)(i32x4*)(void*)g, (AS3 i32x4*)(i32x4*)l, 0, 0);
#else
    *(float4*)l = *(const float4*)g;
#endif
}

__device__ __forceinline__ void wait_async_lds() {
#if __has_builtin(__builtin_amdgcn_global_load_async_to_lds_b128)
#if __has_builtin(__builtin_amdgcn_s_wait_asynccnt)
    __builtin_amdgcn_s_wait_asynccnt(0);
#else
    asm volatile("s_wait_asynccnt 0" ::: "memory");
#endif
#endif
}

// ---------------- argmax over each heatmap (first-index ties) ----------------
__global__ __launch_bounds__(256) void argmax_kernel(const float* __restrict__ hm,
                                                     int* __restrict__ rows,
                                                     int* __restrict__ cols) {
    __shared__ float sv[256];
    __shared__ int   si[256];
    int b = blockIdx.x, t = threadIdx.x;
    const float* p = hm + (size_t)b * NPIX;
    float best = -3.0e38f; int bi = 0;
    for (int i = t; i < NPIX; i += 256) {
        float v = p[i];
        if (v > best) { best = v; bi = i; }
    }
    sv[t] = best; si[t] = bi;
    __syncthreads();
    for (int st = 128; st > 0; st >>= 1) {
        if (t < st) {
            float ov = sv[t + st]; int oi = si[t + st];
            if (ov > sv[t] || (ov == sv[t] && oi < si[t])) { sv[t] = ov; si[t] = oi; }
        }
        __syncthreads();
    }
    if (t == 0) { rows[b] = si[0] / WW; cols[b] = si[0] % WW; }
}

// -------- fused edge-pad crop(32x32 @argmax) + bilinear resize -> 72x96 ------
__global__ __launch_bounds__(256) void resize_kernel(const float* __restrict__ frames,
                                                     const int* __restrict__ rows,
                                                     const int* __restrict__ cols,
                                                     float* __restrict__ out) {
    int b = blockIdx.y;
    int p = blockIdx.x * 256 + threadIdx.x;   // 27*256 = 6912
    int oy = p / WW, ox = p % WW;
    int r = rows[b], c = cols[b];
    const float* fb = frames + (size_t)b * NPIX;
    float sy = (oy + 0.5f) * (32.0f / 72.0f) - 0.5f;
    float sx = (ox + 0.5f) * (32.0f / 96.0f) - 0.5f;
    float y0f = floorf(sy), x0f = floorf(sx);
    float wy = sy - y0f, wx = sx - x0f;
    int y0 = (int)y0f, x0 = (int)x0f;
    auto tap = [&](int i, int j) -> float {
        int ci = iclamp(i, 0, 31), cj = iclamp(j, 0, 31);
        int ri = iclamp(r + ci - 16, 0, HH - 1);
        int rj = iclamp(c + cj - 16, 0, WW - 1);
        return fb[ri * WW + rj];
    };
    float v00 = tap(y0, x0),     v01 = tap(y0, x0 + 1);
    float v10 = tap(y0 + 1, x0), v11 = tap(y0 + 1, x0 + 1);
    float vt = v00 + wx * (v01 - v00);
    float vb = v10 + wx * (v11 - v10);
    out[(size_t)b * NPIX + p] = vt + wy * (vb - vt);
}

// ------- conv1 3x3 (1->32) + bias + BN + ReLU -> f16 NHWC (ic innermost) -----
__global__ __launch_bounds__(256) void conv1_kernel(const float* __restrict__ x,
                                                    const float* __restrict__ w1,
                                                    const float* __restrict__ b1,
                                                    const float* __restrict__ g1,
                                                    const float* __restrict__ bb1,
                                                    const float* __restrict__ m1,
                                                    const float* __restrict__ v1,
                                                    _Float16* __restrict__ act1,
                                                    int b0) {
    __shared__ float sw[32 * 9];
    __shared__ float sscale[32], sshift[32];
    int t = threadIdx.x;
    for (int i = t; i < 288; i += 256) sw[i] = w1[i];
    if (t < 32) {
        float sc = g1[t] * rsqrtf(v1[t] + EPS);
        sscale[t] = sc;
        sshift[t] = (b1[t] - m1[t]) * sc + bb1[t];
    }
    __syncthreads();
    int bl = blockIdx.y;                     // local batch in chunk
    int p = blockIdx.x * 256 + t;            // pixel
    int h = p / WW, wc = p % WW;
    const float* xb = x + (size_t)(b0 + bl) * NPIX;
    float xin[9];
#pragma unroll
    for (int s = 0; s < 9; ++s) {
        int hh = h + s / 3 - 1, ww = wc + s % 3 - 1;
        xin[s] = ((unsigned)hh < (unsigned)HH && (unsigned)ww < (unsigned)WW)
                     ? xb[hh * WW + ww] : 0.0f;
    }
    v16h o0 = {}, o1 = {};
#pragma unroll
    for (int oc = 0; oc < 32; ++oc) {
        float a = 0.0f;
#pragma unroll
        for (int s = 0; s < 9; ++s) a = fmaf(sw[oc * 9 + s], xin[s], a);
        a = a * sscale[oc] + sshift[oc];
        a = a > 0.0f ? a : 0.0f;
        if (oc < 16) o0[oc] = (_Float16)a; else o1[oc - 16] = (_Float16)a;
    }
    _Float16* dst = act1 + ((size_t)bl * NPIX + p) * 32;
    *(v16h*)dst        = o0;
    *(v16h*)(dst + 16) = o1;
}

// ---- pack conv2 weights (BN scale folded) into WMMA A-fragment lane order ---
// layout: [ob(4)][s(9)][lane(32)][j(16)] f16 ; K per step = ic (32)
__global__ void pack_w2_kernel(const float* __restrict__ w2,
                               const float* __restrict__ g2,
                               const float* __restrict__ v2,
                               _Float16* __restrict__ wpack) {
    int tid = blockIdx.x * 256 + threadIdx.x;      // 4*9*32*16 = 18432
    if (tid >= 18432) return;
    int j    = tid & 15;
    int lane = (tid >> 4) & 31;
    int s    = (tid >> 9) % 9;
    int ob   = tid / 4608;
    int oc   = ob * 16 + (lane & 15);
    int ic   = (lane < 16) ? j : j + 16;
    float sc = g2[oc] * rsqrtf(v2[oc] + EPS);
    wpack[tid] = (_Float16)(w2[(oc * 32 + ic) * 9 + s] * sc);
}

__global__ void zero_kernel(float* p, int n) {
    int i = blockIdx.x * 256 + threadIdx.x;
    if (i < n) p[i] = 0.0f;
}

// ---- conv2 implicit GEMM via v_wmma_f32_16x16x32_f16, fused BN/ReLU/pool ----
// Per block: stage 4x18x32ch halo tile in LDS (async global->LDS where
// available), then 8 waves x 9 K-steps of WMMA fed from ds_load.
// tile: 64 oc x [rows 2y,2y+1] x [cols 16x .. 16x+15]
__global__ __launch_bounds__(256) void conv2_pool_kernel(const _Float16* __restrict__ act1,
                                                         const _Float16* __restrict__ wpack,
                                                         const float* __restrict__ c2b,
                                                         const float* __restrict__ g2,
                                                         const float* __restrict__ b2,
                                                         const float* __restrict__ m2,
                                                         const float* __restrict__ v2,
                                                         float* __restrict__ pooled,
                                                         int b0) {
    __shared__ __align__(64) _Float16 stage[4 * 18 * 32];   // 4608 B halo tile
    __shared__ float smem[64 * 2 * 16];
    __shared__ float sh_shift[64];
    int t = threadIdx.x;
    if (t < 64) {
        float sc = g2[t] * rsqrtf(v2[t] + EPS);
        sh_shift[t] = (c2b[t] - m2[t]) * sc + b2[t];
    }
    int bl = blockIdx.z;
    const _Float16* xb = act1 + (size_t)bl * NPIX * 32;
    int gy0 = 2 * blockIdx.y - 1;
    int gx0 = 16 * blockIdx.x - 1;
    // stage layout: [row(4)][col(18)][32ch] f16 ; 288 chunks of 16 B
#pragma unroll
    for (int it = 0; it < 2; ++it) {
        int q = t + it * 256;
        if (q < 288) {
            int row  = q / 72;
            int rem  = q - row * 72;
            int colq = rem >> 2;
            int part = rem & 3;
            int gh = gy0 + row, gw = gx0 + colq;
            char* sp = (char*)stage + (size_t)q * 16;
            if ((unsigned)gh < (unsigned)HH && (unsigned)gw < (unsigned)WW) {
                const _Float16* gp = xb + ((gh * WW + gw) * 32 + part * 8);
                async_copy16(gp, sp);
            } else {
                float4 z = make_float4(0.0f, 0.0f, 0.0f, 0.0f);
                *(float4*)sp = z;   // zero-pad halo (SAME conv padding)
            }
        }
    }
    wait_async_lds();
    __syncthreads();

    int lane = t & 31, w = t >> 5;
    int ob = w & 3, half = w >> 2;
    int icbase = (lane < 16) ? 0 : 16;
    int nloc = lane & 15;
    const _Float16* abase = wpack + (size_t)ob * 4608 + lane * 16;
    v8f acc = {};
#pragma unroll
    for (int s = 0; s < 9; ++s) {
        v16h a = *(const v16h*)(abase + s * 512);
        int row_local = half + s / 3;        // (h+dh) - gy0
        int col_local = nloc + s % 3;        // (col+dw) - gx0
        v16h bfrag = *(const v16h*)(stage + (row_local * 18 + col_local) * 32 + icbase);
        acc = __builtin_amdgcn_wmma_f32_16x16x32_f16(false, a, false, bfrag,
                                                     (short)0, acc, false, false);
    }
    // epilogue: shift + ReLU -> LDS tile [oc][half][n]
    int mofs = (lane < 16) ? 0 : 8;
#pragma unroll
    for (int r = 0; r < 8; ++r) {
        int oc = ob * 16 + r + mofs;
        float vv = acc[r] + sh_shift[oc];
        vv = vv > 0.0f ? vv : 0.0f;
        smem[(oc * 2 + half) * 16 + nloc] = vv;
    }
    __syncthreads();
    if (t < 64) {
        int oc = t;
        float s = 0.0f;
#pragma unroll
        for (int pc = 0; pc < 8; ++pc) {
            float a0 = smem[(oc * 2 + 0) * 16 + 2 * pc];
            float a1 = smem[(oc * 2 + 0) * 16 + 2 * pc + 1];
            float a2 = smem[(oc * 2 + 1) * 16 + 2 * pc];
            float a3 = smem[(oc * 2 + 1) * 16 + 2 * pc + 1];
            s += fmaxf(fmaxf(a0, a1), fmaxf(a2, a3));
        }
        atomicAdd(&pooled[(size_t)(b0 + bl) * 64 + oc], s);
    }
}

// ---- pooled f32 sums -> mean -> f16 for FC1 B-fragments ----
__global__ void cvt_pooled_kernel(const float* __restrict__ pooled, _Float16* __restrict__ ph) {
    int i = blockIdx.x * 256 + threadIdx.x;  // 256*64
    ph[i] = (_Float16)(pooled[i] * (1.0f / 1728.0f));   // 36*48 pooled cells
}

__global__ void cvt_fc1w_kernel(const float* __restrict__ w, _Float16* __restrict__ wh) {
    int i = blockIdx.x * 256 + threadIdx.x;  // 256*64
    wh[i] = (_Float16)w[i];
}

// ---- FC1 (M=256 out, N=256 batch, K=64) via WMMA + BN3 + ReLU ----
__global__ __launch_bounds__(256) void fc1_wmma_kernel(const _Float16* __restrict__ ph,   // [256][64]
                                                       const _Float16* __restrict__ wh,   // [256][64]
                                                       const float* __restrict__ fb,
                                                       const float* __restrict__ g3,
                                                       const float* __restrict__ b3,
                                                       const float* __restrict__ m3,
                                                       const float* __restrict__ v3,
                                                       float* __restrict__ fc1act) {     // [256 batch][256]
    int t = threadIdx.x, lane = t & 31, w = t >> 5;
    int tile = blockIdx.x * 8 + w;           // 32 blocks * 8 waves = 256 tiles
    int mt = tile >> 4, nt = tile & 15;
    int mrow  = mt * 16 + (lane & 15);
    int kofs  = (lane < 16) ? 0 : 16;
    int batch = nt * 16 + (lane & 15);
    int icb   = (lane < 16) ? 0 : 16;
    v8f acc = {};
#pragma unroll
    for (int s = 0; s < 2; ++s) {
        const _Float16* ap = wh + mrow * 64 + 32 * s;
        v8h a0 = *(const v8h*)(ap + kofs);
        v8h a1 = *(const v8h*)(ap + 8 + kofs);
        v16h a;
#pragma unroll
        for (int j = 0; j < 8; ++j) { a[j] = a0[j]; a[j + 8] = a1[j]; }
        v16h bfrag = *(const v16h*)(ph + batch * 64 + 32 * s + icb);
        acc = __builtin_amdgcn_wmma_f32_16x16x32_f16(false, a, false, bfrag,
                                                     (short)0, acc, false, false);
    }
    int mbase = mt * 16 + ((lane < 16) ? 0 : 8);
#pragma unroll
    for (int r = 0; r < 8; ++r) {
        int m = mbase + r;
        float sc = g3[m] * rsqrtf(v3[m] + EPS);
        float val = (acc[r] + fb[m] - m3[m]) * sc + b3[m];
        val = val > 0.0f ? val : 0.0f;
        fc1act[(size_t)batch * 256 + m] = val;
    }
}

// ---- FC2 (256->1) + sigmoid ----
__global__ __launch_bounds__(256) void fc2_kernel(const float* __restrict__ fc1act,
                                                  const float* __restrict__ w2,
                                                  const float* __restrict__ b2s,
                                                  float* __restrict__ out) {
    __shared__ float sw[256];
    int t = threadIdx.x;
    sw[t] = w2[t];
    __syncthreads();
    float a = 0.0f;
    const float* row = fc1act + (size_t)t * 256;
    for (int i = 0; i < 256; ++i) a = fmaf(row[i], sw[i], a);
    a += b2s[0];
    out[t] = 1.0f / (1.0f + __expf(-a));
}

extern "C" void kernel_launch(void* const* d_in, const int* in_sizes, int n_in,
                              void* d_out, int out_size, void* d_ws, size_t ws_size,
                              hipStream_t stream) {
    (void)in_sizes; (void)n_in; (void)out_size; (void)ws_size;
    const float* heatmap = (const float*)d_in[0];
    const float* frames  = (const float*)d_in[1];

    char* ws = (char*)d_ws;
    int*      rows   = (int*)(ws + 0);
    int*      cols   = (int*)(ws + 1024);
    float*    crops  = (float*)(ws + 2048);            // 256*6912 f32
    _Float16* act1   = (_Float16*)(ws + 7079936);      // CHUNK*6912*32 f16
    _Float16* wpack  = (_Float16*)(ws + 21235712);     // 64*288 f16
    float*    pooled = (float*)(ws + 21272576);        // 256*64 f32
    _Float16* ph     = (_Float16*)(ws + 21338112);     // 256*64 f16
    _Float16* fwh    = (_Float16*)(ws + 21370880);     // 256*64 f16
    float*    fc1act = (float*)(ws + 21403648);        // 256*256 f32

    argmax_kernel<<<NB, 256, 0, stream>>>(heatmap, rows, cols);
    resize_kernel<<<dim3(27, NB), 256, 0, stream>>>(frames, rows, cols, crops);

    for (int br = 0; br < 2; ++br) {
        int base = 2 + br * 20;   // param dict insertion order
        const float* conv1_w = (const float*)d_in[base + 0];
        const float* conv1_b = (const float*)d_in[base + 1];
        const float* bn1_g   = (const float*)d_in[base + 2];
        const float* bn1_b   = (const float*)d_in[base + 3];
        const float* bn1_m   = (const float*)d_in[base + 4];
        const float* bn1_v   = (const float*)d_in[base + 5];
        const float* conv2_w = (const float*)d_in[base + 6];
        const float* conv2_b = (const float*)d_in[base + 7];
        const float* bn2_g   = (const float*)d_in[base + 8];
        const float* bn2_b   = (const float*)d_in[base + 9];
        const float* bn2_m   = (const float*)d_in[base + 10];
        const float* bn2_v   = (const float*)d_in[base + 11];
        const float* fc1_w   = (const float*)d_in[base + 12];
        const float* fc1_b   = (const float*)d_in[base + 13];
        const float* bn3_g   = (const float*)d_in[base + 14];
        const float* bn3_b   = (const float*)d_in[base + 15];
        const float* bn3_m   = (const float*)d_in[base + 16];
        const float* bn3_v   = (const float*)d_in[base + 17];
        const float* fc2_w   = (const float*)d_in[base + 18];
        const float* fc2_b   = (const float*)d_in[base + 19];

        const float* x = (br == 0) ? crops : frames;   // click uses crops, hold uses frames

        pack_w2_kernel<<<72, 256, 0, stream>>>(conv2_w, bn2_g, bn2_v, wpack);
        cvt_fc1w_kernel<<<64, 256, 0, stream>>>(fc1_w, fwh);
        zero_kernel<<<64, 256, 0, stream>>>(pooled, NB * 64);

        for (int b0 = 0; b0 < NB; b0 += CHUNK) {
            conv1_kernel<<<dim3(27, CHUNK), 256, 0, stream>>>(
                x, conv1_w, conv1_b, bn1_g, bn1_b, bn1_m, bn1_v, act1, b0);
            conv2_pool_kernel<<<dim3(6, 36, CHUNK), 256, 0, stream>>>(
                act1, wpack, conv2_b, bn2_g, bn2_b, bn2_m, bn2_v, pooled, b0);
        }
        cvt_pooled_kernel<<<64, 256, 0, stream>>>(pooled, ph);
        fc1_wmma_kernel<<<32, 256, 0, stream>>>(ph, fwh, fc1_b, bn3_g, bn3_b, bn3_m, bn3_v, fc1act);
        fc2_kernel<<<1, 256, 0, stream>>>(fc1act, fc2_w, fc2_b, (float*)d_out + br * NB);
    }
}
